// UDE_41987600285704
// MI455X (gfx1250) — matrix operand order repaired and checked by reading, not compiled, not run on hardware
//
#include <hip/hip_runtime.h>
#include <hip/hip_bf16.h>
#include <stdint.h>

typedef _Float16     v16h __attribute__((ext_vector_type(16)));
typedef float        v8f  __attribute__((ext_vector_type(8)));
typedef unsigned int v8u  __attribute__((ext_vector_type(8)));

#define WAVES_PER_BLK 8
#define BLK 256

__device__ __forceinline__ void lds_fence() {
    __asm__ volatile("s_wait_dscnt 0" ::: "memory");
}

// tanh(x + b) with bias pre-scaled: bb = K2E * b, K2E = 2*log2(e).
// tanh(y) = 1 - 2/(e^{2y}+1); exp2 saturation gives exact +-1 tails, no clamp.
#define K2E 2.8853900817779268f
__device__ __forceinline__ float tanh_b(float x, float bb) {
    float e = __builtin_amdgcn_exp2f(__builtin_fmaf(x, K2E, bb));
    return 1.f - 2.f * __builtin_amdgcn_rcpf(e + 1.f);
}

struct WaveCtx {
    _Float16* h;       // 32 traj x 16 halves (staging / transpose buffer)
    float*    f;       // 32 traj x 4 f32 (MLP output transpose-back)
    v16h B1, B2, B3;   // resident WMMA B operands (weights, f16)
    float bb1, bb2, bb3;  // per-lane pre-scaled biases
    int   lane;
};

// Evaluate the 4->16->8->4 tanh MLP for 32 trajectories (2 WMMA tiles,
// interleaved per layer so the two independent WMMAs cover each other's
// WMMA->VALU hazard slots).
__device__ __forceinline__ void mlp_eval(WaveCtx& cx,
        float y0, float y1, float y2, float y3, float out[4])
{
    const int      lane = cx.lane;
    const int      lo   = lane & 15;
    const bool     hi   = lane >= 16;
    const uint32_t msk  = hi ? 0u : 0xFFFFFFFFu;
    const v8f      cz   = {};

    // Stage per-lane state as 4 f16 (one b64 store).
    {
        union { _Float16 hh[4]; uint2 u; } yu;
        yu.hh[0] = (_Float16)y0; yu.hh[1] = (_Float16)y1;
        yu.hh[2] = (_Float16)y2; yu.hh[3] = (_Float16)y3;
        *(uint2*)(cx.h + lane * 16) = yu.u;
    }
    lds_fence();

    // ---- layer 1: [16,4] @ W1[4,16]  (K padded 4 -> 32) ----
    // A layout: lanes<16 elems0..3 = K0..3 (own 4 state values); lanes>=16 zero.
    v8u a0 = {}, a1 = {};
    {
        uint2 w0 = *(const uint2*)(cx.h + lo * 16);          // tile0 rows 0..15
        uint2 w1 = *(const uint2*)(cx.h + (16 + lo) * 16);   // tile1 rows 16..31
        a0[0] = w0.x & msk; a0[1] = w0.y & msk;
        a1[0] = w1.x & msk; a1[1] = w1.y & msk;
    }
    v8f d0 = __builtin_amdgcn_wmma_f32_16x16x32_f16(
        false, __builtin_bit_cast(v16h, a0), false, cx.B1, (short)0, cz, false, false);
    v8f d1 = __builtin_amdgcn_wmma_f32_16x16x32_f16(
        false, __builtin_bit_cast(v16h, a1), false, cx.B1, (short)0, cz, false, false);
    #pragma unroll
    for (int r = 0; r < 8; ++r) {
        int m = hi ? (r + 8) : r;                            // D layout row
        cx.h[m * 16 + lo]        = (_Float16)tanh_b(d0[r], cx.bb1);
        cx.h[(16 + m) * 16 + lo] = (_Float16)tanh_b(d1[r], cx.bb1);
    }
    lds_fence();

    // ---- layer 2: [16,16] @ W2[16,8]  (K padded 16 -> 32) ----
    // A layout: lanes<16 elems0..7 = K0..7 ; lanes>=16 elems0..7 = K8..15.
    v8u a2 = {}, a3 = {};
    {
        const int off = hi ? 8 : 0;
        uint4 w0 = *(const uint4*)(cx.h + lo * 16 + off);
        uint4 w1 = *(const uint4*)(cx.h + (16 + lo) * 16 + off);
        a2[0] = w0.x; a2[1] = w0.y; a2[2] = w0.z; a2[3] = w0.w;
        a3[0] = w1.x; a3[1] = w1.y; a3[2] = w1.z; a3[3] = w1.w;
    }
    d0 = __builtin_amdgcn_wmma_f32_16x16x32_f16(
        false, __builtin_bit_cast(v16h, a2), false, cx.B2, (short)0, cz, false, false);
    d1 = __builtin_amdgcn_wmma_f32_16x16x32_f16(
        false, __builtin_bit_cast(v16h, a3), false, cx.B2, (short)0, cz, false, false);
    if (lo < 8) {
        #pragma unroll
        for (int r = 0; r < 8; ++r) {
            int m = hi ? (r + 8) : r;
            cx.h[m * 16 + lo]        = (_Float16)tanh_b(d0[r], cx.bb2);
            cx.h[(16 + m) * 16 + lo] = (_Float16)tanh_b(d1[r], cx.bb2);
        }
    }
    lds_fence();

    // ---- layer 3: [16,8] @ W3[8,4]  (K padded 8 -> 32) ----
    v8u a4 = {}, a5 = {};
    {
        uint4 w0 = *(const uint4*)(cx.h + lo * 16);
        uint4 w1 = *(const uint4*)(cx.h + (16 + lo) * 16);
        a4[0] = w0.x & msk; a4[1] = w0.y & msk; a4[2] = w0.z & msk; a4[3] = w0.w & msk;
        a5[0] = w1.x & msk; a5[1] = w1.y & msk; a5[2] = w1.z & msk; a5[3] = w1.w & msk;
    }
    d0 = __builtin_amdgcn_wmma_f32_16x16x32_f16(
        false, __builtin_bit_cast(v16h, a4), false, cx.B3, (short)0, cz, false, false);
    d1 = __builtin_amdgcn_wmma_f32_16x16x32_f16(
        false, __builtin_bit_cast(v16h, a5), false, cx.B3, (short)0, cz, false, false);
    if (lo < 4) {
        #pragma unroll
        for (int r = 0; r < 8; ++r) {
            int m = hi ? (r + 8) : r;
            cx.f[m * 4 + lo]        = tanh_b(d0[r], cx.bb3);
            cx.f[(16 + m) * 4 + lo] = tanh_b(d1[r], cx.bb3);
        }
    }
    lds_fence();

    float4 o4 = *(const float4*)(cx.f + lane * 4);
    out[0] = o4.x; out[1] = o4.y; out[2] = o4.z; out[3] = o4.w;
}

__device__ __forceinline__ float fid_rate(float chi, float inv_p15, float e) {
    float t = __builtin_fmaf(e, __cosf(chi), 1.f);
    return t * t * inv_p15;
}

__global__ void __launch_bounds__(BLK, 1)
ude_kernel(const float* __restrict__ ic,
           const float* __restrict__ W1, const float* __restrict__ b1,
           const float* __restrict__ W2, const float* __restrict__ b2,
           const float* __restrict__ W3, const float* __restrict__ b3,
           const int*   __restrict__ nsp,
           float*       __restrict__ out)
{
    __shared__ __align__(16) _Float16 hbuf[WAVES_PER_BLK][32 * 16];
    __shared__ __align__(16) float    fbuf[WAVES_PER_BLK][32 * 4];

    const int lane = threadIdx.x & 31;
    const int wv   = threadIdx.x >> 5;
    const int traj = blockIdx.x * BLK + threadIdx.x;
    const int T    = nsp[0];
    const int TC   = T - 2;

    const float dt   = 10000.0f / 999.0f;
    const float hdt  = 0.5f * dt;
    const float dt6  = dt / 6.0f;
    const float Qf   = 0.01f;
    const float QFAC = Qf / (1.f + Qf);
    const float CK   = 1.5853309190424045f;   // sqrt(4*pi/5)
    const float IDT2 = 1.0f / (dt * dt);

    // ---------------- build resident WMMA operands ----------------
    WaveCtx cx;
    cx.lane = lane;
    cx.h = &hbuf[wv][0];
    cx.f = &fbuf[wv][0];
    cx.B1 = {}; cx.B2 = {}; cx.B3 = {};
    const int n = lane & 15;
    if (n < 8) {   // B2 layout: lanes 0-15 elems0..7 = K0..7, lanes 16-31 elems0..7 = K8..15
        #pragma unroll
        for (int k = 0; k < 8; ++k) cx.B2[k] = (_Float16)W2[(k + ((lane >= 16) ? 8 : 0)) * 8 + n];
    }
    if (lane < 16) {   // B1/B3 layout: lanes 0-15 hold K0.. for column N=lane
        #pragma unroll
        for (int k = 0; k < 4; ++k)  cx.B1[k] = (_Float16)W1[k * 16 + n];
        if (n < 4) {
            #pragma unroll
            for (int k = 0; k < 8; ++k)  cx.B3[k] = (_Float16)W3[k * 4 + n];
        }
    }
    cx.bb1 = K2E * b1[n];
    cx.bb2 = K2E * ((n < 8) ? b2[n] : 0.f);
    cx.bb3 = K2E * ((n < 4) ? b3[n] : 0.f);

    // ---------------- initial state ----------------
    float phi = ic[traj * 4 + 0];
    float chi = ic[traj * 4 + 1];
    float p   = ic[traj * 4 + 2];
    float e   = ic[traj * 4 + 3];

    // Jxx - Jyy = r^2 * cos(2*phi) * Q/(1+Q)   (trace terms cancel)
    float r0  = p * __builtin_amdgcn_rcpf(__builtin_fmaf(e, __cosf(chi), 1.f));
    float jm1 = r0 * r0 * __cosf(2.f * phi) * QFAC;  // Jd(t-1)
    float jm2 = 0.f;                                  // Jd(t-2)
    float rm1 = r0;                                   // r(t-1)

    float* orow = out + (size_t)traj * (size_t)TC;

    #pragma unroll 1
    for (int t = 1; t < T; ++t) {
        // ---- RK4 on the fiducial term (p,e rates are zero) ----
        float ip15 = __frsqrt_rn(p * p * p);
        float d1 = fid_rate(chi, ip15, e);
        float d2 = fid_rate(chi + hdt * d1, ip15, e);
        float d3 = fid_rate(chi + hdt * d2, ip15, e);
        float d4 = fid_rate(chi + dt  * d3, ip15, e);
        float kf = dt6 * (d1 + 2.f * d2 + 2.f * d3 + d4);

        // ---- RK4 on the MLP term (WMMA) ----
        float k1[4], k2[4], k3[4], k4[4];
        mlp_eval(cx, phi, chi, p, e, k1);
        mlp_eval(cx, phi + hdt * k1[0], chi + hdt * k1[1],
                     p   + hdt * k1[2], e   + hdt * k1[3], k2);
        mlp_eval(cx, phi + hdt * k2[0], chi + hdt * k2[1],
                     p   + hdt * k2[2], e   + hdt * k2[3], k3);
        mlp_eval(cx, phi + dt * k3[0], chi + dt * k3[1],
                     p   + dt * k3[2], e   + dt * k3[3], k4);

        phi += kf + dt6 * (k1[0] + 2.f * k2[0] + 2.f * k3[0] + k4[0]);
        chi += kf + dt6 * (k1[1] + 2.f * k2[1] + 2.f * k3[1] + k4[1]);
        p   +=      dt6 * (k1[2] + 2.f * k2[2] + 2.f * k3[2] + k4[2]);
        e   +=      dt6 * (k1[3] + 2.f * k2[3] + 2.f * k3[3] + k4[3]);

        // ---- quadrupole readout, emitted on the fly ----
        float rt  = p * __builtin_amdgcn_rcpf(__builtin_fmaf(e, __cosf(chi), 1.f));
        float jdt = rt * rt * __cosf(2.f * phi) * QFAC;
        if (t >= 2) {
            orow[t - 2] = CK * __builtin_amdgcn_rcpf(rm1)
                          * (jdt - 2.f * jm1 + jm2) * IDT2;
        }
        jm2 = jm1; jm1 = jdt; rm1 = rt;
    }
}

extern "C" void kernel_launch(void* const* d_in, const int* in_sizes, int n_in,
                              void* d_out, int out_size, void* d_ws, size_t ws_size,
                              hipStream_t stream) {
    const float* ic = (const float*)d_in[0];
    const float* W1 = (const float*)d_in[1];
    const float* b1 = (const float*)d_in[2];
    const float* W2 = (const float*)d_in[3];
    const float* b2 = (const float*)d_in[4];
    const float* W3 = (const float*)d_in[5];
    const float* b3 = (const float*)d_in[6];
    const int*   ns = (const int*)d_in[7];
    float* o = (float*)d_out;

    int B = in_sizes[0] / 4;
    int blocks = (B + BLK - 1) / BLK;
    hipLaunchKernelGGL(ude_kernel, dim3(blocks), dim3(BLK), 0, stream,
                       ic, W1, b1, W2, b2, W3, b3, ns, o);
}
